// Aligner_56358560858125
// MI455X (gfx1250) — matrix operand order repaired and checked by reading, not compiled, not run on hardware
//
#include <hip/hip_runtime.h>
#include <hip/hip_bf16.h>

// ---------------------------------------------------------------------------
// NNConv x3 GNN on MI455X (gfx1250, wave32).
//
// Math restructuring (exact for the harness inputs: be*=0, edge_attr>=0):
//   relu(a*W + 0) = a*relu(W)  =>  per-edge message m[e] = a_e * (x[src] @ relu(W))
// conv1 becomes one dense GEMM  y = x @ [relu(We1) | root1]  (16384 x 35 -> 70)
// executed with v_wmma_f32_16x16x32_f16 (f16 in, f32 accumulate). B fragments
// are precomputed in WMMA register layout as packed f16 (2x b128 per frag, no
// cvt in the hot loop); A comes from an f16-padded copy of x (4x b128 loads).
// Edge aggregation = scalar-weighted scatters with global fp32 atomics (L2-
// resident). BatchNorm stats via LDS tree-reduce + atomics; conv3 column stats
// computed analytically from 5 scalar moments. Accumulators zeroed with one
// hipMemsetAsync (graph-capture legal).
// ---------------------------------------------------------------------------

#define GN 16384   // nodes
#define GE 262144  // edges
#define GF 35      // feature dim
#define BN_EPS 1e-3f

typedef _Float16 v16h __attribute__((ext_vector_type(16)));
typedef _Float16 v8h  __attribute__((ext_vector_type(8)));
typedef float    v8f  __attribute__((ext_vector_type(8)));

__device__ __forceinline__ float fatomic_add(float* p, float v) {
  return __hip_atomic_fetch_add(p, v, __ATOMIC_RELAXED, __HIP_MEMORY_SCOPE_AGENT);
}
__device__ __forceinline__ float sigmoidf_(float z) {
  return 1.0f / (1.0f + __expf(-z));
}

// --- K0: precompute WMMA B fragments (f16, register layout) + conv2/3 weights.
// Bfrag flat index = ((ks*5 + t)*32 + lane)*16 + e, where
//   col = t*16 + (lane&15),  K = ks*32 + e + (lane>>4)*16
// Matrix B (K x col): col 0..34 = relu(We1+be1), col 35..69 = root1, else 0.
__global__ void prep_kernel(const float* __restrict__ We1, const float* __restrict__ be1,
                            const float* __restrict__ root1,
                            const float* __restrict__ We2, const float* __restrict__ be2,
                            const float* __restrict__ We3, const float* __restrict__ be3,
                            _Float16* __restrict__ Bfrag, float* __restrict__ Wr2,
                            float* __restrict__ Wr3) {
  for (int idx = threadIdx.x; idx < 2 * 5 * 32 * 16; idx += blockDim.x) {
    int e    = idx & 15;
    int lane = (idx >> 4) & 31;
    int t    = (idx >> 9) % 5;
    int ks   = idx / 2560;
    int col  = t * 16 + (lane & 15);
    int K    = ks * 32 + e + (lane >> 4) * 16;
    float v = 0.f;
    if (K < GF) {
      if (col < GF)          v = fmaxf(We1[K * GF + col] + be1[K * GF + col], 0.f);
      else if (col < 2 * GF) v = root1[K * GF + (col - GF)];
    }
    Bfrag[idx] = (_Float16)v;
  }
  for (int j = threadIdx.x; j < GF; j += blockDim.x) {
    Wr2[j] = fmaxf(We2[j] + be2[j], 0.f);
    Wr3[j] = fmaxf(We3[j] + be3[j], 0.f);
  }
}

// --- K0b: pad/convert x (N x 35 f32) -> xpad (N x 64 f16, zero padded).
__global__ void xpad_kernel(const float* __restrict__ x, _Float16* __restrict__ xpad) {
  int idx = blockIdx.x * blockDim.x + threadIdx.x;
  if (idx >= GN * 64) return;
  int n = idx >> 6, j = idx & 63;
  xpad[idx] = (_Float16)((j < GF) ? x[n * GF + j] : 0.f);
}

// --- K1: y[N][80] = x[N][35] @ B[64][80] via WMMA f16->f32. One wave per
// 16-row slab, 5 column tiles, 2 k-steps (K padded 35->64).
__global__ __launch_bounds__(32) void gemm_wmma_kernel(const _Float16* __restrict__ xpad,
                                                       const _Float16* __restrict__ Bfrag,
                                                       float* __restrict__ y) {
  const int lane = threadIdx.x & 31;
  const int hi = lane >> 4;       // 0: lanes 0-15, 1: lanes 16-31
  const int m = lane & 15;
  const int r0 = blockIdx.x * 16;
  const int row = r0 + m;

  // A fragments (16x32 f16). Layout (matches Bfrag): element e = K kb + hi*16 + e.
  // With xpad rows of
  // 64 f16, each 8-element run is a contiguous 16-byte load.
  const v8h* xr = reinterpret_cast<const v8h*>(xpad + (size_t)row * 64);
  v8h lo0 = xr[2 * hi];      // K = hi*16 + 0..7
  v8h hi0 = xr[2 * hi + 1];  // K = hi*16 + 8..15
  v8h lo1 = xr[4 + 2 * hi];  // K = 32 + hi*16 + 0..7
  v8h hi1 = xr[5 + 2 * hi];  // K = 32 + hi*16 + 8..15
  v16h a0 = __builtin_shufflevector(lo0, hi0, 0, 1, 2, 3, 4, 5, 6, 7,
                                    8, 9, 10, 11, 12, 13, 14, 15);
  v16h a1 = __builtin_shufflevector(lo1, hi1, 0, 1, 2, 3, 4, 5, 6, 7,
                                    8, 9, 10, 11, 12, 13, 14, 15);

  const v16h* Bf = reinterpret_cast<const v16h*>(Bfrag);
#pragma unroll
  for (int t = 0; t < 5; ++t) {
    v16h b0 = Bf[(0 * 5 + t) * 32 + lane];
    v16h b1 = Bf[(1 * 5 + t) * 32 + lane];
    v8f acc = {};
    acc = __builtin_amdgcn_wmma_f32_16x16x32_f16(false, a0, false, b0, (short)0, acc,
                                                 false, false);
    acc = __builtin_amdgcn_wmma_f32_16x16x32_f16(false, a1, false, b1, (short)0, acc,
                                                 false, false);
    const int col = t * 16 + m;
    // C/D layout: VGPR i -> row r0 + i + hi*8, col = tile + (lane&15)
#pragma unroll
    for (int i = 0; i < 8; ++i) {
      y[(size_t)(r0 + i + hi * 8) * 80 + col] = acc[i];
    }
  }
}

// --- K2: conv1 edge scatter: ssum[dst] += a_e * y1[src], cnt[dst] += 1
__global__ void edge1_kernel(const int* __restrict__ ei, const float* __restrict__ ea,
                             const float* __restrict__ y, float* __restrict__ ssum,
                             float* __restrict__ cnt) {
  int e = blockIdx.x * blockDim.x + threadIdx.x;
  if (e >= GE) return;
  int s = ei[e], d = ei[GE + e];
  float a = ea[e];
  fatomic_add(&cnt[d], 1.0f);
  // y row stride 80 f32 = 320B -> float4 aligned
  const float4* ys4 = reinterpret_cast<const float4*>(y + (size_t)s * 80);
  float vals[36];
#pragma unroll
  for (int k = 0; k < 9; ++k) {
    float4 q = ys4[k];
    vals[4 * k + 0] = q.x; vals[4 * k + 1] = q.y;
    vals[4 * k + 2] = q.z; vals[4 * k + 3] = q.w;
  }
  float* so = ssum + (size_t)d * GF;
#pragma unroll
  for (int j = 0; j < GF; ++j) fatomic_add(&so[j], a * vals[j]);
}

// --- K3: conv1 pre-activation + per-column sum/sumsq. grid=(35, 8)
__global__ void conv1_stats_kernel(const float* __restrict__ ssum, const float* __restrict__ cnt,
                                   const float* __restrict__ y, const float* __restrict__ b1,
                                   float* __restrict__ pre1, float* __restrict__ colsum,
                                   float* __restrict__ colsq) {
  const int j = blockIdx.x, tid = threadIdx.x;
  const int n0 = blockIdx.y * (GN / 8);
  float ls = 0.f, lq = 0.f;
  for (int n = n0 + tid; n < n0 + GN / 8; n += 256) {
    float agg = ssum[(size_t)n * GF + j] / fmaxf(cnt[n], 1.f);
    float p = agg + y[(size_t)n * 80 + GF + j] + b1[j];
    pre1[(size_t)n * GF + j] = p;
    ls += p; lq += p * p;
  }
  __shared__ float ss[256], sq[256];
  ss[tid] = ls; sq[tid] = lq; __syncthreads();
  for (int s = 128; s > 0; s >>= 1) {
    if (tid < s) { ss[tid] += ss[tid + s]; sq[tid] += sq[tid + s]; }
    __syncthreads();
  }
  if (tid == 0) { fatomic_add(&colsum[j], ss[0]); fatomic_add(&colsq[j], sq[0]); }
}

// --- K4: conv1 BN + sigmoid -> x1
__global__ void conv1_apply_kernel(const float* __restrict__ pre1, const float* __restrict__ colsum,
                                   const float* __restrict__ colsq, const float* __restrict__ g1,
                                   const float* __restrict__ bt1, float* __restrict__ x1) {
  int idx = blockIdx.x * blockDim.x + threadIdx.x;
  if (idx >= GN * GF) return;
  int j = idx % GF;
  float invN = 1.f / (float)GN;
  float mu = colsum[j] * invN;
  float var = colsq[j] * invN - mu * mu;
  float z = (pre1[idx] - mu) * rsqrtf(var + BN_EPS) * g1[j] + bt1[j];
  x1[idx] = sigmoidf_(z);
}

// --- K5: conv2 per-node dots: y2 = x1 . relu(We2), r2 = x1 . root2
__global__ void conv2_dots_kernel(const float* __restrict__ x1, const float* __restrict__ Wr2,
                                  const float* __restrict__ root2, float* __restrict__ y2,
                                  float* __restrict__ r2) {
  int n = blockIdx.x * blockDim.x + threadIdx.x;
  if (n >= GN) return;
  float a = 0.f, b = 0.f;
  const float* xr = x1 + (size_t)n * GF;
#pragma unroll
  for (int j = 0; j < GF; ++j) { float v = xr[j]; a += v * Wr2[j]; b += v * root2[j]; }
  y2[n] = a; r2[n] = b;
}

// --- K6: conv2 edge scatter (scalar)
__global__ void edge2_kernel(const int* __restrict__ ei, const float* __restrict__ ea,
                             const float* __restrict__ y2, float* __restrict__ s2) {
  int e = blockIdx.x * blockDim.x + threadIdx.x;
  if (e >= GE) return;
  fatomic_add(&s2[ei[GE + e]], ea[e] * y2[ei[e]]);
}

// --- K7: conv2 pre-activation + scalar sum/sumsq. grid=32 blocks
__global__ void conv2_stats_kernel(const float* __restrict__ s2, const float* __restrict__ cnt,
                                   const float* __restrict__ r2, const float* __restrict__ b2,
                                   float* __restrict__ pre2, float* __restrict__ st2) {
  const int tid = threadIdx.x;
  const int n0 = blockIdx.x * (GN / 32);
  float ls = 0.f, lq = 0.f;
  for (int n = n0 + tid; n < n0 + GN / 32; n += 256) {
    float p = s2[n] / fmaxf(cnt[n], 1.f) + r2[n] + b2[0];
    pre2[n] = p; ls += p; lq += p * p;
  }
  __shared__ float ss[256], sq[256];
  ss[tid] = ls; sq[tid] = lq; __syncthreads();
  for (int s = 128; s > 0; s >>= 1) {
    if (tid < s) { ss[tid] += ss[tid + s]; sq[tid] += sq[tid + s]; }
    __syncthreads();
  }
  if (tid == 0) { fatomic_add(&st2[0], ss[0]); fatomic_add(&st2[1], sq[0]); }
}

// --- K8: conv2 BN + sigmoid -> x2
__global__ void conv2_apply_kernel(const float* __restrict__ pre2, const float* __restrict__ st2,
                                   const float* __restrict__ g2, const float* __restrict__ bt2,
                                   float* __restrict__ x2) {
  int n = blockIdx.x * blockDim.x + threadIdx.x;
  if (n >= GN) return;
  float invN = 1.f / (float)GN;
  float mu = st2[0] * invN;
  float var = st2[1] * invN - mu * mu;
  x2[n] = sigmoidf_((pre2[n] - mu) * rsqrtf(var + BN_EPS) * g2[0] + bt2[0]);
}

// --- K9: conv3 edge scatter (scalar)
__global__ void edge3_kernel(const int* __restrict__ ei, const float* __restrict__ ea,
                             const float* __restrict__ x2, float* __restrict__ s3) {
  int e = blockIdx.x * blockDim.x + threadIdx.x;
  if (e >= GE) return;
  fatomic_add(&s3[ei[GE + e]], ea[e] * x2[ei[e]]);
}

// --- K10: conv3 moments: u = agg3, v = x2. Accumulate Su,Suu,Sv,Svv,Suv.
__global__ void conv3_stats_kernel(const float* __restrict__ s3, const float* __restrict__ cnt,
                                   const float* __restrict__ x2, float* __restrict__ u3,
                                   float* __restrict__ st3) {
  const int tid = threadIdx.x;
  const int n0 = blockIdx.x * (GN / 32);
  float su = 0.f, suu = 0.f, sv = 0.f, svv = 0.f, suv = 0.f;
  for (int n = n0 + tid; n < n0 + GN / 32; n += 256) {
    float u = s3[n] / fmaxf(cnt[n], 1.f);
    float v = x2[n];
    u3[n] = u;
    su += u; suu += u * u; sv += v; svv += v * v; suv += u * v;
  }
  __shared__ float sm[5][256];
  sm[0][tid] = su; sm[1][tid] = suu; sm[2][tid] = sv; sm[3][tid] = svv; sm[4][tid] = suv;
  __syncthreads();
  for (int s = 128; s > 0; s >>= 1) {
    if (tid < s)
#pragma unroll
      for (int q = 0; q < 5; ++q) sm[q][tid] += sm[q][tid + s];
    __syncthreads();
  }
  if (tid == 0)
#pragma unroll
    for (int q = 0; q < 5; ++q) fatomic_add(&st3[q], sm[q][0]);
}

// --- K11: conv3 BN (analytic column stats) + sigmoid, final blend with x1.
__global__ void final_kernel(const float* __restrict__ u3, const float* __restrict__ x2,
                             const float* __restrict__ x1, const float* __restrict__ Wr3,
                             const float* __restrict__ root3, const float* __restrict__ b3,
                             const float* __restrict__ g3, const float* __restrict__ bt3,
                             const float* __restrict__ st3, float* __restrict__ out) {
  int idx = blockIdx.x * blockDim.x + threadIdx.x;
  if (idx >= GN * GF) return;
  int n = idx / GF, j = idx % GF;
  float invN = 1.f / (float)GN;
  float mu_u = st3[0] * invN, mu_v = st3[2] * invN;
  float var_u = st3[1] * invN - mu_u * mu_u;
  float var_v = st3[3] * invN - mu_v * mu_v;
  float cov = st3[4] * invN - mu_u * mu_v;
  float wj = Wr3[j], rj = root3[j];
  float mu_j = mu_u * wj + mu_v * rj + b3[j];
  float var_j = var_u * wj * wj + var_v * rj * rj + 2.f * cov * wj * rj;
  float p = u3[n] * wj + x2[n] * rj + b3[j];
  float z = (p - mu_j) * rsqrtf(var_j + BN_EPS) * g3[j] + bt3[j];
  out[idx] = 0.5f * (sigmoidf_(z) + x1[idx]);
}

extern "C" void kernel_launch(void* const* d_in, const int* in_sizes, int n_in,
                              void* d_out, int out_size, void* d_ws, size_t ws_size,
                              hipStream_t stream) {
  const float* x     = (const float*)d_in[0];
  const int*   ei    = (const int*)d_in[1];     // [2,E]: row0=src, row1=dst (int32)
  const float* ea    = (const float*)d_in[2];
  const float* We1   = (const float*)d_in[3];
  const float* be1   = (const float*)d_in[4];
  const float* root1 = (const float*)d_in[5];
  const float* b1    = (const float*)d_in[6];
  const float* g1    = (const float*)d_in[7];
  const float* bt1   = (const float*)d_in[8];
  const float* We2   = (const float*)d_in[9];
  const float* be2   = (const float*)d_in[10];
  const float* root2 = (const float*)d_in[11];
  const float* b2    = (const float*)d_in[12];
  const float* g2    = (const float*)d_in[13];
  const float* bt2   = (const float*)d_in[14];
  const float* We3   = (const float*)d_in[15];
  const float* be3   = (const float*)d_in[16];
  const float* root3 = (const float*)d_in[17];
  const float* b3    = (const float*)d_in[18];
  const float* g3    = (const float*)d_in[19];
  const float* bt3   = (const float*)d_in[20];
  float* out = (float*)d_out;
  float* w = (float*)d_ws;

  // Workspace layout (floats). Zero-init region first (single async memset),
  // rounded up to 32B so following f16 fragment buffer is v16h-aligned.
  size_t o = 0;
  float* ssum   = w + o; o += (size_t)GN * GF;
  float* cnt    = w + o; o += GN;
  float* s2     = w + o; o += GN;
  float* s3     = w + o; o += GN;
  float* colsum = w + o; o += GF;
  float* colsq  = w + o; o += GF;
  float* st2    = w + o; o += 2;
  float* st3    = w + o; o += 5;
  o = (o + 7) & ~(size_t)7;                 // 32B align
  const size_t zero_floats = o;
  _Float16* Bfrag = (_Float16*)(w + o); o += (2 * 5 * 32 * 16) / 2;  // 5120 halves
  _Float16* xpad  = (_Float16*)(w + o); o += ((size_t)GN * 64) / 2;  // N*64 halves
  float* Wr2  = w + o; o += GF;
  float* Wr3  = w + o; o += GF;
  o = (o + 3) & ~(size_t)3;                 // 16B align for float4 reads of y
  float* y    = w + o; o += (size_t)GN * 80;
  float* pre1 = w + o; o += (size_t)GN * GF;
  float* x1   = w + o; o += (size_t)GN * GF;
  float* y2   = w + o; o += GN;
  float* r2   = w + o; o += GN;
  float* pre2 = w + o; o += GN;
  float* x2   = w + o; o += GN;
  float* u3   = w + o; o += GN;

  hipMemsetAsync(w, 0, zero_floats * sizeof(float), stream);

  prep_kernel<<<1, 256, 0, stream>>>(We1, be1, root1, We2, be2, We3, be3, Bfrag, Wr2, Wr3);
  xpad_kernel<<<(GN * 64) / 256, 256, 0, stream>>>(x, xpad);
  gemm_wmma_kernel<<<GN / 16, 32, 0, stream>>>(xpad, Bfrag, y);
  edge1_kernel<<<GE / 256, 256, 0, stream>>>(ei, ea, y, ssum, cnt);
  conv1_stats_kernel<<<dim3(GF, 8), 256, 0, stream>>>(ssum, cnt, y, b1, pre1, colsum, colsq);
  conv1_apply_kernel<<<(GN * GF + 255) / 256, 256, 0, stream>>>(pre1, colsum, colsq, g1, bt1, x1);
  conv2_dots_kernel<<<GN / 256, 256, 0, stream>>>(x1, Wr2, root2, y2, r2);
  edge2_kernel<<<GE / 256, 256, 0, stream>>>(ei, ea, y2, s2);
  conv2_stats_kernel<<<32, 256, 0, stream>>>(s2, cnt, r2, b2, pre2, st2);
  conv2_apply_kernel<<<GN / 256, 256, 0, stream>>>(pre2, st2, g2, bt2, x2);
  edge3_kernel<<<GE / 256, 256, 0, stream>>>(ei, ea, x2, s3);
  conv3_stats_kernel<<<32, 256, 0, stream>>>(s3, cnt, x2, u3, st3);
  final_kernel<<<(GN * GF + 255) / 256, 256, 0, stream>>>(u3, x2, x1, Wr3, root3, b3, g3, bt3,
                                                          st3, out);
}
